// Transformer_36481452212853
// MI455X (gfx1250) — matrix operand order repaired and checked by reading, not compiled, not run on hardware
//
#include <hip/hip_runtime.h>
#include <hip/hip_bf16.h>

typedef __attribute__((ext_vector_type(16))) _Float16 v16h;
typedef __attribute__((ext_vector_type(8)))  _Float16 v8h;
typedef __attribute__((ext_vector_type(8)))  float    v8f;

#define D_DIM 128
#define H_DIM 8
#define DH_DIM 16

// ---------------- elementwise / conversion kernels ----------------

__global__ void k_f32_to_f16(const float* __restrict__ x, _Float16* __restrict__ y, long n) {
  long i = (long)blockIdx.x * blockDim.x + threadIdx.x;
  if (i < n) y[i] = (_Float16)x[i];
}

__global__ void k_zero(float* __restrict__ p, long n) {
  long i = (long)blockIdx.x * blockDim.x + threadIdx.x;
  if (i < n) p[i] = 0.0f;
}

// W: row-major [K x Nc] f32  ->  Wt: row-major [Nc x K] f16 (i.e. col-major B)
__global__ void k_conv_transpose(const float* __restrict__ W, _Float16* __restrict__ Wt,
                                 int K, int Nc) {
  long i = (long)blockIdx.x * blockDim.x + threadIdx.x;
  if (i >= (long)K * Nc) return;
  int k = (int)(i / Nc);
  int n = (int)(i % Nc);
  Wt[(long)n * K + k] = (_Float16)W[i];
}

// ---------------- WMMA GEMM: C[MxNc] = A[MxK](f16) * Bt[NcxK](f16,col-major) + bias ----------------
// block = 128 threads (4 waves), tile 64(M) x 128(N), K-step 32.
// Tiles staged with GLOBAL_LOAD_ASYNC_TO_LDS_B128 (ASYNCcnt), double-buffered;
// all fragments preloaded and pinned by a scheduling fence so the 8 WMMAs issue
// as a dependency-free burst behind a single s_wait_dscnt.

#define TM 64
#define TN 128
#define TK 32
#define LDA 40   // halves; 80B row pitch (16B aligned, conflict-padded)
#define LDB 40

__device__ __forceinline__ void async_b128(unsigned lds_off, const _Float16* g) {
  asm volatile("global_load_async_to_lds_b128 %0, %1, off"
               :: "v"(lds_off), "v"((unsigned long long)(uintptr_t)g) : "memory");
}
__device__ __forceinline__ void async_b128_o16(unsigned lds_off, const _Float16* g) {
  asm volatile("global_load_async_to_lds_b128 %0, %1, off offset:16"
               :: "v"(lds_off), "v"((unsigned long long)(uintptr_t)g) : "memory");
}
__device__ __forceinline__ void async_b128_o32(unsigned lds_off, const _Float16* g) {
  asm volatile("global_load_async_to_lds_b128 %0, %1, off offset:32"
               :: "v"(lds_off), "v"((unsigned long long)(uintptr_t)g) : "memory");
}
__device__ __forceinline__ void async_b128_o48(unsigned lds_off, const _Float16* g) {
  asm volatile("global_load_async_to_lds_b128 %0, %1, off offset:48"
               :: "v"(lds_off), "v"((unsigned long long)(uintptr_t)g) : "memory");
}

__global__ __launch_bounds__(128) void k_gemm_wmma(
    const _Float16* __restrict__ A, const _Float16* __restrict__ Bt,
    const float* __restrict__ bias, float* __restrict__ Cf,
    _Float16* __restrict__ Ch, int M, int K, int Nc, int relu)
{
  __shared__ _Float16 sA[2][TM * LDA];
  __shared__ _Float16 sB[2][TN * LDB];

  const int t    = threadIdx.x;
  const int wave = t >> 5;
  const int lane = t & 31;
  const int m0   = blockIdx.x * TM;
  const int n0   = blockIdx.y * TN;

  v8f acc[8] = {};

  // staging assignments (all lanes active; A rows past M are clamped — they only
  // ever feed output rows past M, which are store-guarded)
  const int arow = t >> 1;          // 0..63
  const int acol = (t & 1) * 16;    // 0 or 16 (halves)
  int gRow = m0 + arow;
  if (gRow >= M) gRow = M - 1;
  const int bcol = t;               // 0..127

  const _Float16* Abase = A + (long)gRow * K + acol;          // +k0 per tile
  const _Float16* Bbase = Bt + (long)(n0 + bcol) * K;         // +k0 per tile

  const unsigned sAoff0 = (unsigned)(uintptr_t)&sA[0][arow * LDA + acol];
  const unsigned sAoff1 = (unsigned)(uintptr_t)&sA[1][arow * LDA + acol];
  const unsigned sBoff0 = (unsigned)(uintptr_t)&sB[0][bcol * LDB];
  const unsigned sBoff1 = (unsigned)(uintptr_t)&sB[1][bcol * LDB];

  // 6 async b128 transfers per thread per tile (2 for A, 4 for B)
  auto issue_tile = [&](int k0, int buf) {
    const unsigned sa = buf ? sAoff1 : sAoff0;
    const unsigned sb = buf ? sBoff1 : sBoff0;
    const _Float16* ga = Abase + k0;
    const _Float16* gb = Bbase + k0;
    async_b128    (sa, ga);
    async_b128_o16(sa, ga);
    async_b128    (sb, gb);
    async_b128_o16(sb, gb);
    async_b128_o32(sb, gb);
    async_b128_o48(sb, gb);
  };

  const int nk = K / TK;
  issue_tile(0, 0);

  for (int i = 0; i < nk; i++) {
    const int buf = i & 1;
    if (i + 1 < nk) {
      issue_tile((i + 1) * TK, (i + 1) & 1);   // prefetch next tile into other buffer
      if (i + 2 < nk)
        __builtin_prefetch(Abase + (i + 2) * TK, 0, 3);  // global_prefetch_b8 (L2 warm)
      asm volatile("s_wait_asynccnt 0x6" ::: "memory");  // tile i done (in-order), 6 in flight
    } else {
      asm volatile("s_wait_asynccnt 0x0" ::: "memory");
    }
    __syncthreads();   // tile i visible to all waves

    // ---- A fragment (ISA 16-bit A 16x32 layout) ----
    const int ar = wave * 16 + (lane & 15);
    const int kb = (lane >> 4) * 8;   // K base 0 or 8; second half at +16
  const int kr = (lane >> 4) * 8;   // B: same as A, K base 0 or 8; second half at +16
    const _Float16* sAp = &sA[buf][ar * LDA + kb];
    v8h alo = *(const v8h*)sAp;
    v8h ahi = *(const v8h*)(sAp + 16);
    v16h afrag;
#pragma unroll
    for (int q = 0; q < 8; q++) { afrag[q] = alo[q]; afrag[8 + q] = ahi[q]; }

    // ---- preload all 8 B fragments (batched ds_loads) ----
    v16h bfrag[8];
#pragma unroll
    for (int nt = 0; nt < 8; nt++) {
      const int bc = nt * 16 + (lane & 15);
      const _Float16* sBp = &sB[buf][bc * LDB + kr];
      v8h blo = *(const v8h*)sBp;
      v8h bhi = *(const v8h*)(sBp + 16);
#pragma unroll
      for (int q = 0; q < 8; q++) { bfrag[nt][q] = blo[q]; bfrag[nt][8 + q] = bhi[q]; }
    }

    // Scheduling fence: LDS loads above cannot be sunk past this point, so the
    // compiler emits all 18 ds_load_b128, ONE s_wait_dscnt, then the WMMA burst.
    asm volatile("" ::: "memory");

    // ---- 8 back-to-back WMMAs: distinct accumulators, shared A, no D->A/B hazard ----
#pragma unroll
    for (int nt = 0; nt < 8; nt++) {
      acc[nt] = __builtin_amdgcn_wmma_f32_16x16x32_f16(
          false, afrag, false, bfrag[nt], (short)0, acc[nt], false, false);
    }
    __syncthreads();   // all waves done with buf before it is re-filled (iter i+2)
  }

  // ---- writeback: C/D layout: lanes0-15 VGPRj -> M=j ; lanes16-31 -> M=8+j ----
  const int rbase = m0 + wave * 16 + ((lane >> 4) ? 8 : 0);
  const int cl = lane & 15;
#pragma unroll
  for (int nt = 0; nt < 8; nt++) {
    const int col = n0 + nt * 16 + cl;
    const float bv = bias ? bias[col] : 0.0f;
#pragma unroll
    for (int j = 0; j < 8; j++) {
      const int row = rbase + j;
      if (row < M) {
        float v = acc[nt][j] + bv;
        if (relu) v = v > 0.0f ? v : 0.0f;
        if (Cf) Cf[(long)row * Nc + col] = v;
        if (Ch) Ch[(long)row * Nc + col] = (_Float16)v;
      }
    }
  }
}

// ---------------- edge kernels ----------------

__global__ void k_edge_scores(const float* __restrict__ kf, const float* __restrict__ qf,
                              const int* __restrict__ src, const int* __restrict__ dst,
                              float* __restrict__ sexp, int E)
{
  long idx = (long)blockIdx.x * blockDim.x + threadIdx.x;
  if (idx >= (long)E * H_DIM) return;
  int e = (int)(idx >> 3), h = (int)(idx & 7);
  const float4* kp = (const float4*)(kf + (long)src[e] * D_DIM + h * DH_DIM);
  const float4* qp = (const float4*)(qf + (long)dst[e] * D_DIM + h * DH_DIM);
  float d = 0.0f;
#pragma unroll
  for (int i = 0; i < 4; i++) {
    float4 a = kp[i], b = qp[i];
    d += a.x * b.x + a.y * b.y + a.z * b.z + a.w * b.w;
  }
  d *= 0.25f;                        // 1/sqrt(16)
  d = fminf(5.0f, fmaxf(-5.0f, d));
  sexp[idx] = __expf(d);
}

__global__ void k_scatter(const float* __restrict__ vf, const int* __restrict__ src,
                          const int* __restrict__ dst, const float* __restrict__ sexp,
                          float* __restrict__ wV, float* __restrict__ z, int E)
{
  long idx = (long)blockIdx.x * blockDim.x + threadIdx.x;
  if (idx >= (long)E * H_DIM) return;
  int e = (int)(idx >> 3), h = (int)(idx & 7);
  float s = sexp[idx];
  int sN = src[e], dN = dst[e];
  const float* vp = vf + (long)sN * D_DIM + h * DH_DIM;
  float* wp = wV + (long)dN * D_DIM + h * DH_DIM;
#pragma unroll
  for (int i = 0; i < DH_DIM; i++) atomicAdd(&wp[i], vp[i] * s);
  atomicAdd(&z[(long)dN * H_DIM + h], s);
}

__global__ void k_attn_div(const float* __restrict__ wV, const float* __restrict__ z,
                           _Float16* __restrict__ attn16, long n)
{
  long i = (long)blockIdx.x * blockDim.x + threadIdx.x;
  if (i >= n) return;
  long node = i >> 7;
  int h = (int)((i & 127) >> 4);
  attn16[i] = (_Float16)(wV[i] / (z[node * H_DIM + h] + 1e-6f));
}

// ---------------- fused residual + layernorm (one wave per row, wave32) ----------------

__global__ __launch_bounds__(128) void k_residual_ln(
    const float* __restrict__ hin, const float* __restrict__ o,
    const float* __restrict__ w, const float* __restrict__ b,
    float* __restrict__ hout, _Float16* __restrict__ hout16, int Nn)
{
  const int wave = threadIdx.x >> 5;
  const int lane = threadIdx.x & 31;
  const int row  = blockIdx.x * 4 + wave;
  if (row >= Nn) return;
  float x[4];
  float s = 0.0f;
#pragma unroll
  for (int i = 0; i < 4; i++) {
    int c = lane + i * 32;
    x[i] = hin[(long)row * D_DIM + c] + o[(long)row * D_DIM + c];
    s += x[i];
  }
#pragma unroll
  for (int off = 16; off > 0; off >>= 1) s += __shfl_xor(s, off, 32);
  float mean = s * (1.0f / 128.0f);
  float v = 0.0f;
#pragma unroll
  for (int i = 0; i < 4; i++) { float d = x[i] - mean; v += d * d; }
#pragma unroll
  for (int off = 16; off > 0; off >>= 1) v += __shfl_xor(v, off, 32);
  float rstd = rsqrtf(v * (1.0f / 128.0f) + 1e-5f);
#pragma unroll
  for (int i = 0; i < 4; i++) {
    int c = lane + i * 32;
    float y = (x[i] - mean) * rstd * w[c] + b[c];
    hout[(long)row * D_DIM + c] = y;
    if (hout16) hout16[(long)row * D_DIM + c] = (_Float16)y;
  }
}

// ---------------- host orchestration ----------------

static inline size_t align256(size_t x) { return (x + 255) & ~(size_t)255; }

extern "C" void kernel_launch(void* const* d_in, const int* in_sizes, int n_in,
                              void* d_out, int out_size, void* d_ws, size_t ws_size,
                              hipStream_t stream) {
  (void)n_in; (void)out_size; (void)ws_size;
  const float* h0   = (const float*)d_in[0];
  const int*   src  = (const int*)  d_in[1];
  const int*   dst  = (const int*)  d_in[2];
  const float* Wemb = (const float*)d_in[3];
  const float* Wq   = (const float*)d_in[4];
  const float* bq   = (const float*)d_in[5];
  const float* Wk   = (const float*)d_in[6];
  const float* bk   = (const float*)d_in[7];
  const float* Wv   = (const float*)d_in[8];
  const float* bv   = (const float*)d_in[9];
  const float* Wo   = (const float*)d_in[10];
  const float* bo   = (const float*)d_in[11];
  const float* ln1w = (const float*)d_in[12];
  const float* ln1b = (const float*)d_in[13];
  const float* Wf1  = (const float*)d_in[14];
  const float* bf1  = (const float*)d_in[15];
  const float* Wf2  = (const float*)d_in[16];
  const float* bf2  = (const float*)d_in[17];
  const float* ln2w = (const float*)d_in[18];
  const float* ln2b = (const float*)d_in[19];

  const int N = in_sizes[0] / D_DIM;
  const int E = in_sizes[1];
  const int L = 2;

  // ---- workspace carve-up ----
  char* p = (char*)d_ws;
  size_t cur = 0;
  auto take = [&](size_t bytes) { void* q = p + cur; cur += align256(bytes); return q; };

  _Float16* h16a   = (_Float16*)take((size_t)N * D_DIM * 2);
  _Float16* h16b   = (_Float16*)take((size_t)N * D_DIM * 2);
  float*    qf     = (float*)   take((size_t)N * D_DIM * 4);
  float*    kf     = (float*)   take((size_t)N * D_DIM * 4);
  float*    vf     = (float*)   take((size_t)N * D_DIM * 4);
  size_t slab = (size_t)E * H_DIM * 4;
  size_t t16b = (size_t)N * 2 * D_DIM * 2;
  float*    sexp   = (float*)   take(slab > t16b ? slab : t16b);   // aliased: sexp | t16
  float*    wV     = (float*)   take((size_t)N * D_DIM * 4);       // aliased: wV | hx
  float*    zacc   = (float*)   take((size_t)N * H_DIM * 4);
  _Float16* attn16 = (_Float16*)take((size_t)N * D_DIM * 2);       // aliased: attn16 | hx16
  float*    hcur   = (float*)   take((size_t)N * D_DIM * 4);
  _Float16* WembT  = (_Float16*)take((size_t)D_DIM * D_DIM * 2);
  _Float16* WqT[2], *WkT[2], *WvT[2], *WoT[2], *Wf1T[2], *Wf2T[2];
  for (int l = 0; l < L; l++) {
    WqT[l]  = (_Float16*)take((size_t)D_DIM * D_DIM * 2);
    WkT[l]  = (_Float16*)take((size_t)D_DIM * D_DIM * 2);
    WvT[l]  = (_Float16*)take((size_t)D_DIM * D_DIM * 2);
    WoT[l]  = (_Float16*)take((size_t)D_DIM * D_DIM * 2);
    Wf1T[l] = (_Float16*)take((size_t)D_DIM * 2 * D_DIM * 2);
    Wf2T[l] = (_Float16*)take((size_t)D_DIM * 2 * D_DIM * 2);
  }
  // aliases
  float*    obuf  = qf;                 // o written after q is dead
  float*    fbuf  = kf;                 // ffn2 out after k is dead
  float*    hx    = wV;                 // hx written after wV consumed
  _Float16* hx16  = attn16;             // hx16 written after attn16 consumed
  _Float16* t16   = (_Float16*)sexp;    // ffn hidden after sexp consumed

  const int DD = D_DIM * D_DIM;

  // ---- weight conversion (transpose to col-major f16) ----
  {
    int nEl = DD, blk = 256, g = (nEl + blk - 1) / blk;
    k_conv_transpose<<<g, blk, 0, stream>>>(Wemb, WembT, D_DIM, D_DIM);
    for (int l = 0; l < L; l++) {
      k_conv_transpose<<<g, blk, 0, stream>>>(Wq + (size_t)l * DD, WqT[l], D_DIM, D_DIM);
      k_conv_transpose<<<g, blk, 0, stream>>>(Wk + (size_t)l * DD, WkT[l], D_DIM, D_DIM);
      k_conv_transpose<<<g, blk, 0, stream>>>(Wv + (size_t)l * DD, WvT[l], D_DIM, D_DIM);
      k_conv_transpose<<<g, blk, 0, stream>>>(Wo + (size_t)l * DD, WoT[l], D_DIM, D_DIM);
      int g2 = (2 * DD + blk - 1) / blk;
      k_conv_transpose<<<g2, blk, 0, stream>>>(Wf1 + (size_t)l * 2 * DD, Wf1T[l], D_DIM, 2 * D_DIM);
      k_conv_transpose<<<g2, blk, 0, stream>>>(Wf2 + (size_t)l * 2 * DD, Wf2T[l], 2 * D_DIM, D_DIM);
    }
  }

  const long nHD = (long)N * D_DIM;
  const int  blk = 256;
  const dim3 gemm_blk(128);
  const dim3 gemm_g128((N + TM - 1) / TM, 1);
  const dim3 gemm_g256((N + TM - 1) / TM, 2);
  const int  edge_g = (int)(((long)E * H_DIM + blk - 1) / blk);
  const int  ln_g   = (N + 3) / 4;

  // h16a = f16(h0); emb GEMM -> hcur (f32) + h16b (f16)
  k_f32_to_f16<<<(int)((nHD + blk - 1) / blk), blk, 0, stream>>>(h0, h16a, nHD);
  k_gemm_wmma<<<gemm_g128, gemm_blk, 0, stream>>>(h16a, WembT, nullptr, hcur, h16b,
                                                  N, D_DIM, D_DIM, 0);

  float*    cur32 = hcur;
  _Float16* cur16 = h16b;

  for (int l = 0; l < L; l++) {
    // Q,K,V projections
    k_gemm_wmma<<<gemm_g128, gemm_blk, 0, stream>>>(cur16, WqT[l], bq + (size_t)l * D_DIM,
                                                    qf, nullptr, N, D_DIM, D_DIM, 0);
    k_gemm_wmma<<<gemm_g128, gemm_blk, 0, stream>>>(cur16, WkT[l], bk + (size_t)l * D_DIM,
                                                    kf, nullptr, N, D_DIM, D_DIM, 0);
    k_gemm_wmma<<<gemm_g128, gemm_blk, 0, stream>>>(cur16, WvT[l], bv + (size_t)l * D_DIM,
                                                    vf, nullptr, N, D_DIM, D_DIM, 0);
    // edge scores + scatter
    k_edge_scores<<<edge_g, blk, 0, stream>>>(kf, qf, src, dst, sexp, E);
    k_zero<<<(int)((nHD + blk - 1) / blk), blk, 0, stream>>>(wV, nHD);
    k_zero<<<(int)(((long)N * H_DIM + blk - 1) / blk), blk, 0, stream>>>(zacc, (long)N * H_DIM);
    k_scatter<<<edge_g, blk, 0, stream>>>(vf, src, dst, sexp, wV, zacc, E);
    k_attn_div<<<(int)((nHD + blk - 1) / blk), blk, 0, stream>>>(wV, zacc, attn16, nHD);
    // O projection + residual LN1
    k_gemm_wmma<<<gemm_g128, gemm_blk, 0, stream>>>(attn16, WoT[l], bo + (size_t)l * D_DIM,
                                                    obuf, nullptr, N, D_DIM, D_DIM, 0);
    k_residual_ln<<<ln_g, 128, 0, stream>>>(cur32, obuf, ln1w + (size_t)l * D_DIM,
                                            ln1b + (size_t)l * D_DIM, hx, hx16, N);
    // FFN
    k_gemm_wmma<<<gemm_g256, gemm_blk, 0, stream>>>(hx16, Wf1T[l], bf1 + (size_t)l * 2 * D_DIM,
                                                    nullptr, t16, N, D_DIM, 2 * D_DIM, 1);
    k_gemm_wmma<<<gemm_g128, gemm_blk, 0, stream>>>(t16, Wf2T[l], bf2 + (size_t)l * D_DIM,
                                                    fbuf, nullptr, N, 2 * D_DIM, D_DIM, 0);
    // residual LN2 -> next h (final layer writes straight to d_out)
    float* outp = (l == L - 1) ? (float*)d_out : hcur;
    k_residual_ln<<<ln_g, 128, 0, stream>>>(hx, fbuf, ln2w + (size_t)l * D_DIM,
                                            ln2b + (size_t)l * D_DIM, outp, h16a, N);
    cur32 = outp;
    cur16 = h16a;
  }
}
